// PrimitivesEmbeddingDGCNGn_70686571758085
// MI455X (gfx1250) — matrix operand order repaired and checked by reading, not compiled, not run on hardware
//
#include <hip/hip_runtime.h>
#include <hip/hip_bf16.h>
#include <stdint.h>

#define NSN   7000          // number of sub-sampled points
#define NSP   7008          // padded to multiple of 16
#define NTC   (NSP / 16)    // 438 column/row tiles
#define KNN   80

typedef __attribute__((ext_vector_type(16))) _Float16 v16h;
typedef __attribute__((ext_vector_type(8)))  float    v8f;

// ---------------------------------------------------------------------------
// WMMA helpers (CDNA5 wave32, V_WMMA_F32_16X16X32_F16)
// ---------------------------------------------------------------------------
__device__ __forceinline__ v8f wmma16(v16h a, v16h b, v8f c) {
  return __builtin_amdgcn_wmma_f32_16x16x32_f16(
      /*neg_a=*/false, a, /*neg_b=*/false, b,
      /*c_mod=*/(short)0, c, /*reuse_a=*/false, /*reuse_b=*/false);
}

// A fragment: 16x32 f16, A row-major with leading dim lda.
// Lanes 0-15 hold row m=lane, K = {0..7, 16..23}; lanes 16-31: K = {8..15, 24..31}.
__device__ __forceinline__ v16h load_a16(const _Float16* A, int lda) {
  int lane = threadIdx.x & 31;
  int m = lane & 15;
  int kb = (lane < 16) ? 0 : 8;
  const _Float16* p = A + m * lda;
  v16h a;
#pragma unroll
  for (int i = 0; i < 8; ++i) a[i] = p[kb + i];
#pragma unroll
  for (int i = 0; i < 8; ++i) a[8 + i] = p[16 + kb + i];
  return a;
}

// B fragment (32x16) loaded from an N-major (N x K, leading dim ldk) matrix:
// lane<16 -> col n=lane, K = {0..7, 16..23} ; lane>=16 -> col n=lane-16, K = {8..15, 24..31}.
__device__ __forceinline__ v16h load_bt(const _Float16* Bt, int ldk) {
  int lane = threadIdx.x & 31;
  int n = lane & 15;
  const _Float16* p = Bt + n * ldk + ((lane < 16) ? 0 : 8);
  v16h b;
#pragma unroll
  for (int i = 0; i < 8; ++i) { b[i] = p[i]; b[8 + i] = p[16 + i]; }
  return b;
}

__device__ __forceinline__ float wave_sum(float v) {
#pragma unroll
  for (int m = 16; m >= 1; m >>= 1) v += __shfl_xor(v, m, 32);
  return v;
}

// ---------------------------------------------------------------------------
// Utility kernels
// ---------------------------------------------------------------------------
__global__ __launch_bounds__(256) void zero_kernel(uint32_t* p, size_t n) {
  size_t t = (size_t)blockIdx.x * 256 + threadIdx.x;
  if (t < n) p[t] = 0u;
}

// Convert/pad a weight matrix (Oreal x Cin f32) -> (Opad x Cinp f16), zero pad.
__global__ __launch_bounds__(256) void wprep_kernel(const float* __restrict__ w,
                                                    _Float16* __restrict__ wh,
                                                    int Oreal, int Cin, int Cinp,
                                                    int total) {
  int t = blockIdx.x * 256 + threadIdx.x;
  if (t >= total) return;
  int o = t / Cinp, c = t - o * Cinp;
  float v = (o < Oreal && c < Cin) ? w[o * Cin + c] : 0.0f;
  wh[t] = (_Float16)v;
}

// Edge-conv weight split: w (O x 2C) -> w1h = w[:, :C]  and  wdh = w[:, C:] - w[:, :C]
// (y = W1 * x_neighbor + (W2 - W1) * x_center ; second term is k-independent)
__global__ __launch_bounds__(256) void wprep_edge_kernel(const float* __restrict__ w,
                                                         _Float16* __restrict__ w1h,
                                                         _Float16* __restrict__ wdh,
                                                         int C, int Cp, int total) {
  int t = blockIdx.x * 256 + threadIdx.x;
  if (t >= total) return;
  int o = t / Cp, c = t - o * Cp;
  float a = 0.f, d = 0.f;
  if (c < C) { a = w[o * 2 * C + c]; d = w[o * 2 * C + C + c] - a; }
  w1h[t] = (_Float16)a;
  wdh[t] = (_Float16)d;
}

// x0 = points[subidx].T as point-major f16 (NSP x 32), cols 0..2 real.
__global__ __launch_bounds__(256) void prep_points_kernel(const float* __restrict__ points,
                                                          const int* __restrict__ subidx,
                                                          _Float16* __restrict__ xh1) {
  int n = blockIdx.x * 256 + threadIdx.x;
  if (n >= NSN) return;
  int j = subidx[n];
#pragma unroll
  for (int c = 0; c < 3; ++c) xh1[n * 32 + c] = (_Float16)points[j * 3 + c];
}

// squared norms per point (f32) from point-major f16 features
__global__ __launch_bounds__(256) void sqnorm_kernel(const _Float16* __restrict__ xh,
                                                     float* __restrict__ xx,
                                                     int Cp, int Cin) {
  int n = blockIdx.x * 256 + threadIdx.x;
  if (n >= NSP) return;
  float s = 0.f;
  for (int c = 0; c < Cin; ++c) { float v = (float)xh[n * Cp + c]; s += v * v; }
  xx[n] = s;
}

// ---------------------------------------------------------------------------
// Pairwise-distance GEMM: negd[n][m] = 2*<x_n,x_m> - xx[n] - xx[m]  (WMMA)
// ---------------------------------------------------------------------------
__global__ __launch_bounds__(256) void dist_kernel(const _Float16* __restrict__ xh,
                                                   const float* __restrict__ xx,
                                                   float* __restrict__ negd, int Cp) {
  int w = threadIdx.x >> 5;
  int ct = blockIdx.x * 8 + w;          // column tile
  int rt = blockIdx.y;                  // row tile
  if (ct >= NTC) return;                // wave-uniform
  v8f acc = {};
  for (int kk = 0; kk < Cp; kk += 32) {
    v16h a = load_a16(xh + (size_t)(rt * 16) * Cp + kk, Cp);
    v16h b = load_bt(xh + (size_t)(ct * 16) * Cp + kk, Cp);
    acc = wmma16(a, b, acc);
  }
  int lane = threadIdx.x & 31;
  int col = ct * 16 + (lane & 15);
  int rbase = rt * 16 + ((lane >> 4) << 3);
  float xxc = xx[col];
#pragma unroll
  for (int r = 0; r < 8; ++r) {
    int row = rbase + r;
    float v = 2.f * acc[r] - xx[row] - xxc;
    if (col >= NSN) v = -3.0e38f;       // never selected by top-k
    negd[(size_t)row * NSP + col] = v;
  }
}

// ---------------------------------------------------------------------------
// Top-K (K=80) per row via 4-pass radix select in LDS (order of the K indices
// is irrelevant downstream: gather + max-over-k + group stats are symmetric).
// ~5 row passes instead of 80 arg-max passes.
// ---------------------------------------------------------------------------
__global__ __launch_bounds__(256) void topk_kernel(const float* __restrict__ negd,
                                                   int* __restrict__ idx) {
  __shared__ uint32_t keys[NSP];
  __shared__ uint32_t hist[256];
  __shared__ uint32_t sh_prefix, sh_above, sh_remain, sh_thresh;
  __shared__ uint32_t ctr_gt, ctr_eq;
  int n = blockIdx.x;
  int tid = threadIdx.x;
  // monotone float -> uint mapping (bigger float => bigger key)
  for (int j = tid; j < NSP; j += 256) {
    float v = (j < NSN) ? negd[(size_t)n * NSP + j] : -3.0e38f;
    uint32_t b = __float_as_uint(v);
    keys[j] = b ^ ((b & 0x80000000u) ? 0xFFFFFFFFu : 0x80000000u);
  }
  if (tid == 0) { sh_prefix = 0u; sh_above = 0u; sh_remain = KNN; }
  __syncthreads();
  for (int pass = 0; pass < 4; ++pass) {
    int shift = 24 - pass * 8;
    uint32_t mask = (pass == 0) ? 0u : (0xFFFFFFFFu << (shift + 8));
    hist[tid] = 0u;
    __syncthreads();
    uint32_t prefix = sh_prefix;
    for (int j = tid; j < NSP; j += 256) {
      uint32_t k = keys[j];
      if ((k & mask) == prefix) atomicAdd(&hist[(k >> shift) & 0xFFu], 1u);
    }
    __syncthreads();
    if (tid == 0) {
      uint32_t rem = sh_remain;
      uint32_t cum = 0; int b = 255;
      for (; b > 0; --b) {
        uint32_t h = hist[b];
        if (cum + h >= rem) break;
        cum += h;
      }
      sh_above += cum;                       // strictly-greater count so far
      sh_remain = rem - cum;
      sh_prefix = prefix | ((uint32_t)b << shift);
      if (pass == 3) sh_thresh = sh_prefix;
    }
    __syncthreads();
  }
  if (tid == 0) { ctr_gt = 0u; ctr_eq = 0u; }
  __syncthreads();
  uint32_t T = sh_thresh;
  uint32_t nGreater = sh_above;              // count(keys > T)
  uint32_t nTies = KNN - nGreater;
  for (int j = tid; j < NSP; j += 256) {
    uint32_t k = keys[j];
    if (k > T) {
      uint32_t p = atomicAdd(&ctr_gt, 1u);
      idx[n * KNN + p] = j;
    } else if (k == T) {
      uint32_t p = atomicAdd(&ctr_eq, 1u);
      if (p < nTies) idx[n * KNN + nGreater + p] = j;
    }
  }
}

// ---------------------------------------------------------------------------
// d[n][o] = (W2 - W1) * x_n  : small plain WMMA conv, f32 point-major out.
// ---------------------------------------------------------------------------
__global__ __launch_bounds__(256) void convd_kernel(const _Float16* __restrict__ wdh,
                                                    const _Float16* __restrict__ xh,
                                                    int Cp, float* __restrict__ dmat,
                                                    int O) {
  int w = threadIdx.x >> 5;
  if (w >= (O >> 4)) return;            // wave-uniform
  int ot = w;
  int nt = blockIdx.x;
  v8f acc = {};
  for (int kk = 0; kk < Cp; kk += 32) {
    v16h a = load_a16(wdh + (size_t)(ot * 16) * Cp + kk, Cp);
    v16h b = load_bt(xh + (size_t)(nt * 16) * Cp + kk, Cp);
    acc = wmma16(a, b, acc);
  }
  int lane = threadIdx.x & 31;
  int n = nt * 16 + (lane & 15);
#pragma unroll
  for (int r = 0; r < 8; ++r) {
    int o = ot * 16 + r + ((lane >> 4) << 3);
    dmat[(size_t)n * O + o] = acc[r];
  }
}

// ---------------------------------------------------------------------------
// Fused edge-conv: gathered GEMM on raw neighbor rows (per-lane row pointer
// from knn, features L2-resident) + k-independent center term from dmat.
// Running max over k (GN+leaky monotone, gn_w==1>0) + group sum/sumsq atomics.
// Never materializes y (O x NS x 80); no LDS staging needed at all.
// ---------------------------------------------------------------------------
__global__ __launch_bounds__(256) void edge_conv_kernel(const _Float16* __restrict__ xh,
                                                        const _Float16* __restrict__ w1h,
                                                        const int* __restrict__ knn,
                                                        const float* __restrict__ dmat,
                                                        float* __restrict__ ymax,
                                                        float* __restrict__ gacc,
                                                        int Cp, int O, int NPB) {
  int tid = threadIdx.x;
  int w = tid >> 5;
  int wpp = O >> 4;                 // waves per point (4 for O=64, 8 for O=128)
  int pb = w / wpp;
  int ot = w - pb * wpp;
  int n = blockIdx.x * NPB + pb;
  int lane = tid & 31;
  int m = lane & 15;
  int kb = (lane < 16) ? 0 : 8;
  float dval = dmat[(size_t)n * O + ot * 16 + m];   // column term (o = ot*16 + m)
  float colmax = -3.0e38f;
  float lsum = 0.f, lsq = 0.f;
  for (int mt = 0; mt < 5; ++mt) {  // 5 * 16 = 80 neighbor rows
    int j = knn[n * KNN + mt * 16 + m];
    const _Float16* arow = xh + (size_t)j * Cp;
    v8f acc = {};
    for (int kk = 0; kk < Cp; kk += 32) {
      v16h a;
#pragma unroll
      for (int i = 0; i < 8; ++i) a[i] = arow[kk + kb + i];
#pragma unroll
      for (int i = 0; i < 8; ++i) a[8 + i] = arow[kk + 16 + kb + i];
      v16h b = load_bt(w1h + (size_t)(ot * 16) * Cp + kk, Cp);
      acc = wmma16(a, b, acc);
    }
#pragma unroll
    for (int r = 0; r < 8; ++r) {
      float v = acc[r] + dval;
      lsum += v; lsq += v * v;
      colmax = fmaxf(colmax, v);
    }
  }
  colmax = fmaxf(colmax, __shfl_xor(colmax, 16, 32));
  if (lane < 16) ymax[(size_t)n * O + ot * 16 + lane] = colmax;
  lsum = wave_sum(lsum);
  lsq  = wave_sum(lsq);
  if (lane == 0) {
    int g = (ot * 16) / (O >> 1);   // groups = 2
    atomicAdd(&gacc[g], lsum);
    atomicAdd(&gacc[8 + g], lsq);
  }
}

// normalize max-over-k, leaky-relu, emit f16 features (+ slice of concat buffer)
__global__ __launch_bounds__(256) void finalize_edge_kernel(const float* __restrict__ ymax,
                                                            const float* __restrict__ gacc,
                                                            const float* __restrict__ gw,
                                                            const float* __restrict__ gb,
                                                            _Float16* __restrict__ xh_out,
                                                            int outCp,
                                                            _Float16* __restrict__ bigT,
                                                            int bigoff, int O, float invcnt) {
  int t = blockIdx.x * 256 + threadIdx.x;
  if (t >= NSN * O) return;
  int n = t / O, o = t - n * O;
  int g = o / (O >> 1);
  float m  = gacc[g] * invcnt;
  float var = gacc[8 + g] * invcnt - m * m;
  float rs = rsqrtf(var + 1e-5f);
  float v = (ymax[(size_t)n * O + o] - m) * rs * gw[o] + gb[o];
  v = v > 0.f ? v : 0.2f * v;
  xh_out[(size_t)n * outCp + o] = (_Float16)v;
  bigT[(size_t)n * 1280 + bigoff + o] = (_Float16)v;
}

// ---------------------------------------------------------------------------
// mlp1 (256 -> 1024) fused with global max over points + group stats (never
// stores the 1024 x 7000 activation; relu/gn monotone).
// ---------------------------------------------------------------------------
__global__ __launch_bounds__(256) void mlp1_kernel(const _Float16* __restrict__ wh,
                                                   const _Float16* __restrict__ bigT,
                                                   const float* __restrict__ bias,
                                                   float* __restrict__ chmax,
                                                   float* __restrict__ gacc) {
  int w = threadIdx.x >> 5;
  int ot = blockIdx.x * 8 + w;          // 64 row tiles
  int lane = threadIdx.x & 31;
  float rmax[8];
#pragma unroll
  for (int r = 0; r < 8; ++r) rmax[r] = -3.0e38f;
  float lsum = 0.f, lsq = 0.f;
  for (int nt = 0; nt < NTC; ++nt) {
    if (nt + 1 < NTC)
      __builtin_prefetch(bigT + (size_t)(nt + 1) * 16 * 1280 + 1024, 0, 1);
    v8f acc = {};
    for (int kk = 0; kk < 256; kk += 32) {
      v16h a = load_a16(wh + (size_t)(ot * 16) * 256 + kk, 256);
      v16h b = load_bt(bigT + (size_t)(nt * 16) * 1280 + 1024 + kk, 1280);
      acc = wmma16(a, b, acc);
    }
    int n = nt * 16 + (lane & 15);
    if (n < NSN) {
#pragma unroll
      for (int r = 0; r < 8; ++r) {
        int o = ot * 16 + r + ((lane >> 4) << 3);
        float v = acc[r] + bias[o];
        rmax[r] = fmaxf(rmax[r], v);
        lsum += v; lsq += v * v;
      }
    }
  }
#pragma unroll
  for (int r = 0; r < 8; ++r) {
    float v = rmax[r];
#pragma unroll
    for (int m = 1; m < 16; m <<= 1) v = fmaxf(v, __shfl_xor(v, m, 32));
    rmax[r] = v;
  }
  if ((lane & 15) == 0) {
#pragma unroll
    for (int r = 0; r < 8; ++r)
      chmax[ot * 16 + r + ((lane >> 4) << 3)] = rmax[r];
  }
  lsum = wave_sum(lsum);
  lsq  = wave_sum(lsq);
  if (lane == 0) {
    int g = ot >> 3;                    // gsize=128 -> 8 tiles/group
    atomicAdd(&gacc[g], lsum);
    atomicAdd(&gacc[8 + g], lsq);
  }
}

__global__ __launch_bounds__(256) void finalize_mlp_kernel(const float* __restrict__ chmax,
                                                           const float* __restrict__ gacc,
                                                           const float* __restrict__ gw,
                                                           const float* __restrict__ gb,
                                                           float* __restrict__ x4,
                                                           float invcnt) {
  int o = blockIdx.x * 256 + threadIdx.x;
  if (o >= 1024) return;
  int g = o >> 7;
  float m = gacc[g] * invcnt;
  float var = gacc[8 + g] * invcnt - m * m;
  float rs = rsqrtf(var + 1e-5f);
  float v = (chmax[o] - m) * rs * gw[o] + gb[o];
  x4[o] = v > 0.f ? v : 0.f;
}

__global__ __launch_bounds__(256) void bcast_kernel(const float* __restrict__ x4,
                                                    _Float16* __restrict__ bigT) {
  int t = blockIdx.x * 256 + threadIdx.x;
  if (t >= NSN * 1024) return;
  int n = t >> 10, o = t & 1023;
  bigT[(size_t)n * 1280 + o] = (_Float16)x4[o];
}

// ---------------------------------------------------------------------------
// Generic conv1d (O x Cin) with bias + group-stat accumulation (WMMA)
// ---------------------------------------------------------------------------
__global__ __launch_bounds__(256) void conv_gn_kernel(const _Float16* __restrict__ wh,
                                                      const _Float16* __restrict__ feat,
                                                      int ldk, int Kd,
                                                      const float* __restrict__ bias,
                                                      float* __restrict__ y,
                                                      float* __restrict__ gacc,
                                                      int O, int gsize) {
  int w = threadIdx.x >> 5;
  int ot = blockIdx.y * 8 + w;
  int nt = blockIdx.x;
  __builtin_prefetch(feat + (size_t)(nt * 16) * ldk, 0, 1);
  v8f acc = {};
  for (int kk = 0; kk < Kd; kk += 32) {
    v16h a = load_a16(wh + (size_t)(ot * 16) * Kd + kk, Kd);
    v16h b = load_bt(feat + (size_t)(nt * 16) * ldk + kk, ldk);
    acc = wmma16(a, b, acc);
  }
  int lane = threadIdx.x & 31;
  int n = nt * 16 + (lane & 15);
  bool valid = n < NSN;
  float lsum = 0.f, lsq = 0.f;
#pragma unroll
  for (int r = 0; r < 8; ++r) {
    int o = ot * 16 + r + ((lane >> 4) << 3);
    float v = acc[r] + bias[o];
    y[(size_t)n * O + o] = v;
    if (valid) { lsum += v; lsq += v * v; }
  }
  lsum = wave_sum(lsum);
  lsq  = wave_sum(lsq);
  if (lane == 0) {
    int g = (ot * 16) / gsize;
    atomicAdd(&gacc[g], lsum);
    atomicAdd(&gacc[8 + g], lsq);
  }
}

__global__ __launch_bounds__(256) void finalize_gn_relu_kernel(const float* __restrict__ y,
                                                               const float* __restrict__ gacc,
                                                               const float* __restrict__ gw,
                                                               const float* __restrict__ gb,
                                                               _Float16* __restrict__ outh,
                                                               int O, int gsize, float invcnt) {
  int t = blockIdx.x * 256 + threadIdx.x;
  if (t >= NSN * O) return;
  int n = t / O, o = t - n * O;
  int g = o / gsize;
  float m = gacc[g] * invcnt;
  float var = gacc[8 + g] * invcnt - m * m;
  float rs = rsqrtf(var + 1e-5f);
  float v = (y[(size_t)n * O + o] - m) * rs * gw[o] + gb[o];
  outh[(size_t)n * O + o] = (_Float16)(v > 0.f ? v : 0.f);
}

// head conv (no GN): out[n*Oreal + o] = W x + b
__global__ __launch_bounds__(256) void head_kernel(const _Float16* __restrict__ wh,
                                                   const _Float16* __restrict__ feat,
                                                   const float* __restrict__ bias,
                                                   float* __restrict__ out,
                                                   int Oreal, int Opad) {
  int w = threadIdx.x >> 5;
  if (w >= (Opad >> 4)) return;       // wave-uniform
  int ot = w;
  int nt = blockIdx.x;
  v8f acc = {};
  for (int kk = 0; kk < 256; kk += 32) {
    v16h a = load_a16(wh + (size_t)(ot * 16) * 256 + kk, 256);
    v16h b = load_bt(feat + (size_t)(nt * 16) * 256 + kk, 256);
    acc = wmma16(a, b, acc);
  }
  int lane = threadIdx.x & 31;
  int n = nt * 16 + (lane & 15);
#pragma unroll
  for (int r = 0; r < 8; ++r) {
    int o = ot * 16 + r + ((lane >> 4) << 3);
    if (n < NSN && o < Oreal) out[(size_t)n * Oreal + o] = acc[r] + bias[o];
  }
}

__global__ __launch_bounds__(256) void logsoftmax8_kernel(const float* __restrict__ tlog,
                                                          float* __restrict__ out) {
  int n = blockIdx.x * 256 + threadIdx.x;
  if (n >= NSN) return;
  float v[8], mx = -3.0e38f;
#pragma unroll
  for (int i = 0; i < 8; ++i) { v[i] = tlog[n * 8 + i]; mx = fmaxf(mx, v[i]); }
  float s = 0.f;
#pragma unroll
  for (int i = 0; i < 8; ++i) s += expf(v[i] - mx);
  float lse = mx + logf(s);
#pragma unroll
  for (int i = 0; i < 8; ++i) out[n * 8 + i] = v[i] - lse;
}

// ---------------------------------------------------------------------------
// Host orchestration
// ---------------------------------------------------------------------------
extern "C" void kernel_launch(void* const* d_in, const int* in_sizes, int n_in,
                              void* d_out, int out_size, void* d_ws, size_t ws_size,
                              hipStream_t stream) {
  (void)in_sizes; (void)n_in; (void)out_size; (void)ws_size;
  const float* points  = (const float*)d_in[0];
  const int*   subidx  = (const int*)d_in[2];
  const float* conv1_w = (const float*)d_in[3];
  const float* gn1_w   = (const float*)d_in[4];
  const float* gn1_b   = (const float*)d_in[5];
  const float* conv2_w = (const float*)d_in[6];
  const float* gn2_w   = (const float*)d_in[7];
  const float* gn2_b   = (const float*)d_in[8];
  const float* conv3_w = (const float*)d_in[9];
  const float* gn3_w   = (const float*)d_in[10];
  const float* gn3_b   = (const float*)d_in[11];
  const float* mlp1_w  = (const float*)d_in[12];
  const float* mlp1_b  = (const float*)d_in[13];
  const float* bnm_w   = (const float*)d_in[14];
  const float* bnm_b   = (const float*)d_in[15];
  const float* dc1_w   = (const float*)d_in[16];
  const float* dc1_b   = (const float*)d_in[17];
  const float* dbn1_w  = (const float*)d_in[18];
  const float* dbn1_b  = (const float*)d_in[19];
  const float* dc2_w   = (const float*)d_in[20];
  const float* dc2_b   = (const float*)d_in[21];
  const float* dbn2_w  = (const float*)d_in[22];
  const float* dbn2_b  = (const float*)d_in[23];
  const float* seg1_w  = (const float*)d_in[24];
  const float* seg1_b  = (const float*)d_in[25];
  const float* bnseg_w = (const float*)d_in[26];
  const float* bnseg_b = (const float*)d_in[27];
  const float* seg2_w  = (const float*)d_in[28];
  const float* seg2_b  = (const float*)d_in[29];
  const float* prim1_w = (const float*)d_in[30];
  const float* prim1_b = (const float*)d_in[31];
  const float* bnprim_w= (const float*)d_in[32];
  const float* bnprim_b= (const float*)d_in[33];
  const float* prim2_w = (const float*)d_in[34];
  const float* prim2_b = (const float*)d_in[35];
  const float* par1_w  = (const float*)d_in[36];
  const float* par1_b  = (const float*)d_in[37];
  const float* bnpar_w = (const float*)d_in[38];
  const float* bnpar_b = (const float*)d_in[39];
  const float* par2_w  = (const float*)d_in[40];
  const float* par2_b  = (const float*)d_in[41];
  float* out = (float*)d_out;

  char* ws = (char*)d_ws;
  size_t off = 0;
  auto alloc = [&](size_t b) -> char* {
    char* p = ws + off; off += (b + 255) & ~(size_t)255; return p;
  };

  // ---- zero-initialized region (padded rows/cols must be 0) ----
  _Float16* xh1   = (_Float16*)alloc((size_t)NSP * 32 * 2);
  _Float16* xh2   = (_Float16*)alloc((size_t)NSP * 64 * 2);
  _Float16* xh3   = (_Float16*)alloc((size_t)NSP * 64 * 2);
  _Float16* xh4   = (_Float16*)alloc((size_t)NSP * 128 * 2);
  _Float16* bigT  = (_Float16*)alloc((size_t)NSP * 1280 * 2);
  _Float16* y1h   = (_Float16*)alloc((size_t)NSP * 512 * 2);
  _Float16* xallh = (_Float16*)alloc((size_t)NSP * 256 * 2);
  _Float16* eh    = (_Float16*)alloc((size_t)NSP * 256 * 2);
  _Float16* th    = (_Float16*)alloc((size_t)NSP * 256 * 2);
  _Float16* qh    = (_Float16*)alloc((size_t)NSP * 256 * 2);
  _Float16* w1e   = (_Float16*)alloc((size_t)64 * 32 * 2);
  _Float16* wd1e  = (_Float16*)alloc((size_t)64 * 32 * 2);
  _Float16* w2e   = (_Float16*)alloc((size_t)64 * 64 * 2);
  _Float16* wd2e  = (_Float16*)alloc((size_t)64 * 64 * 2);
  _Float16* w3e   = (_Float16*)alloc((size_t)128 * 64 * 2);
  _Float16* wd3e  = (_Float16*)alloc((size_t)128 * 64 * 2);
  _Float16* wm1h  = (_Float16*)alloc((size_t)1024 * 256 * 2);
  _Float16* wd1h  = (_Float16*)alloc((size_t)512 * 1280 * 2);
  _Float16* wd2h  = (_Float16*)alloc((size_t)256 * 512 * 2);
  _Float16* ws1h  = (_Float16*)alloc((size_t)256 * 256 * 2);
  _Float16* ws2h  = (_Float16*)alloc((size_t)64 * 256 * 2);
  _Float16* wp1h  = (_Float16*)alloc((size_t)256 * 256 * 2);
  _Float16* wp2h  = (_Float16*)alloc((size_t)16 * 256 * 2);
  _Float16* wq1h  = (_Float16*)alloc((size_t)256 * 256 * 2);
  _Float16* wq2h  = (_Float16*)alloc((size_t)32 * 256 * 2);
  float* xx    = (float*)alloc((size_t)NSP * 4);
  float* gacc  = (float*)alloc((size_t)16 * 4);
  float* chmax = (float*)alloc((size_t)1024 * 4);
  float* x4f   = (float*)alloc((size_t)1024 * 4);
  size_t zwords = off / 4;

  // ---- write-before-read region (no init needed) ----
  float* ymax = (float*)alloc((size_t)NSP * 128 * 4);
  float* dmat = (float*)alloc((size_t)NSP * 128 * 4);
  float* ybuf = (float*)alloc((size_t)NSP * 512 * 4);
  float* tlog = (float*)alloc((size_t)NSN * 8 * 4);
  int*   knn  = (int*)alloc((size_t)NSN * KNN * 4);
  float* negd = (float*)alloc((size_t)NSP * NSP * 4);

  // ---- init ----
  zero_kernel<<<(int)((zwords + 255) / 256), 256, 0, stream>>>((uint32_t*)ws, zwords);

  auto wprep = [&](const float* w, _Float16* wh, int Oreal, int Opad, int Cin, int Cinp) {
    int total = Opad * Cinp;
    wprep_kernel<<<(total + 255) / 256, 256, 0, stream>>>(w, wh, Oreal, Cin, Cinp, total);
  };
  auto wprep_edge = [&](const float* w, _Float16* w1h, _Float16* wdh, int O, int C, int Cp) {
    int total = O * Cp;
    wprep_edge_kernel<<<(total + 255) / 256, 256, 0, stream>>>(w, w1h, wdh, C, Cp, total);
  };
  wprep_edge(conv1_w, w1e, wd1e, 64, 3, 32);
  wprep_edge(conv2_w, w2e, wd2e, 64, 64, 64);
  wprep_edge(conv3_w, w3e, wd3e, 128, 64, 64);
  wprep(mlp1_w, wm1h, 1024, 1024, 256, 256);
  wprep(dc1_w,  wd1h, 512, 512, 1280, 1280);
  wprep(dc2_w,  wd2h, 256, 256, 512, 512);
  wprep(seg1_w, ws1h, 256, 256, 256, 256);
  wprep(seg2_w, ws2h, 50, 64, 256, 256);
  wprep(prim1_w,wp1h, 256, 256, 256, 256);
  wprep(prim2_w,wp2h, 8, 16, 256, 256);
  wprep(par1_w, wq1h, 256, 256, 256, 256);
  wprep(par2_w, wq2h, 22, 32, 256, 256);

  prep_points_kernel<<<(NSN + 255) / 256, 256, 0, stream>>>(points, subidx, xh1);

  auto zero_gacc = [&]() {
    zero_kernel<<<1, 256, 0, stream>>>((uint32_t*)gacc, 16);
  };

  // ---- edge-conv layer (knn -> center GEMM -> fused gathered GEMM -> finalize) ----
  auto edge_layer = [&](const _Float16* xin, int Cp, int Cin,
                        const _Float16* w1h, const _Float16* wdh, int O,
                        const float* gw, const float* gb, _Float16* xout, int outCp,
                        int bigoff) {
    sqnorm_kernel<<<(NSP + 255) / 256, 256, 0, stream>>>(xin, xx, Cp, Cin);
    dist_kernel<<<dim3((NTC + 7) / 8, NTC), 256, 0, stream>>>(xin, xx, negd, Cp);
    topk_kernel<<<NSN, 256, 0, stream>>>(negd, knn);
    convd_kernel<<<NTC, 256, 0, stream>>>(wdh, xin, Cp, dmat, O);
    zero_gacc();
    int NPB = 8 / (O >> 4);                      // points per block
    edge_conv_kernel<<<NSN / NPB, 256, 0, stream>>>(xin, w1h, knn, dmat, ymax, gacc,
                                                    Cp, O, NPB);
    float invcnt = 1.0f / ((float)(O >> 1) * (float)NSN * (float)KNN);
    finalize_edge_kernel<<<(NSN * O + 255) / 256, 256, 0, stream>>>(
        ymax, gacc, gw, gb, xout, outCp, bigT, bigoff, O, invcnt);
  };

  edge_layer(xh1, 32, 3,  w1e, wd1e, 64,  gn1_w, gn1_b, xh2, 64,  1024);
  edge_layer(xh2, 64, 64, w2e, wd2e, 64,  gn2_w, gn2_b, xh3, 64,  1088);
  edge_layer(xh3, 64, 64, w3e, wd3e, 128, gn3_w, gn3_b, xh4, 128, 1152);

  // ---- mlp1: 256 -> 1024, max over points, broadcast into bigT cols 0..1023 ----
  zero_gacc();
  mlp1_kernel<<<8, 256, 0, stream>>>(wm1h, bigT, mlp1_b, chmax, gacc);
  finalize_mlp_kernel<<<4, 256, 0, stream>>>(chmax, gacc, bnm_w, bnm_b, x4f,
                                             1.0f / (128.0f * (float)NSN));
  bcast_kernel<<<(NSN * 1024 + 255) / 256, 256, 0, stream>>>(x4f, bigT);

  // ---- dc1: 1280 -> 512 ----
  zero_gacc();
  conv_gn_kernel<<<dim3(NTC, 512 / 128), 256, 0, stream>>>(wd1h, bigT, 1280, 1280,
                                                           dc1_b, ybuf, gacc, 512, 64);
  finalize_gn_relu_kernel<<<(NSN * 512 + 255) / 256, 256, 0, stream>>>(
      ybuf, gacc, dbn1_w, dbn1_b, y1h, 512, 64, 1.0f / (64.0f * (float)NSN));

  // ---- dc2: 512 -> 256 ----
  zero_gacc();
  conv_gn_kernel<<<dim3(NTC, 256 / 128), 256, 0, stream>>>(wd2h, y1h, 512, 512,
                                                           dc2_b, ybuf, gacc, 256, 64);
  finalize_gn_relu_kernel<<<(NSN * 256 + 255) / 256, 256, 0, stream>>>(
      ybuf, gacc, dbn2_w, dbn2_b, xallh, 256, 64, 1.0f / (64.0f * (float)NSN));

  // ---- seg branch ----
  zero_gacc();
  conv_gn_kernel<<<dim3(NTC, 2), 256, 0, stream>>>(ws1h, xallh, 256, 256,
                                                   seg1_b, ybuf, gacc, 256, 64);
  finalize_gn_relu_kernel<<<(NSN * 256 + 255) / 256, 256, 0, stream>>>(
      ybuf, gacc, bnseg_w, bnseg_b, eh, 256, 64, 1.0f / (64.0f * (float)NSN));
  head_kernel<<<NTC, 256, 0, stream>>>(ws2h, eh, seg2_b, out, 50, 64);

  // ---- prim branch ----
  zero_gacc();
  conv_gn_kernel<<<dim3(NTC, 2), 256, 0, stream>>>(wp1h, xallh, 256, 256,
                                                   prim1_b, ybuf, gacc, 256, 64);
  finalize_gn_relu_kernel<<<(NSN * 256 + 255) / 256, 256, 0, stream>>>(
      ybuf, gacc, bnprim_w, bnprim_b, th, 256, 64, 1.0f / (64.0f * (float)NSN));
  head_kernel<<<NTC, 256, 0, stream>>>(wp2h, th, prim2_b, tlog, 8, 16);
  logsoftmax8_kernel<<<(NSN + 255) / 256, 256, 0, stream>>>(tlog, out + (size_t)NSN * 50);

  // ---- par branch ----
  zero_gacc();
  conv_gn_kernel<<<dim3(NTC, 2), 256, 0, stream>>>(wq1h, xallh, 256, 256,
                                                   par1_b, ybuf, gacc, 256, 64);
  finalize_gn_relu_kernel<<<(NSN * 256 + 255) / 256, 256, 0, stream>>>(
      ybuf, gacc, bnpar_w, bnpar_b, qh, 256, 64, 1.0f / (64.0f * (float)NSN));
  head_kernel<<<NTC, 256, 0, stream>>>(wq2h, qh, par2_b, out + (size_t)NSN * 58, 22, 32);
}
